// Dereverber_v3_44427141710483
// MI455X (gfx1250) — matrix operand order repaired and checked by reading, not compiled, not run on hardware
//
#include <hip/hip_runtime.h>
#include <hip/hip_bf16.h>
#include <stdint.h>

// ---------------------------------------------------------------------------
// Dims: S samples, M slots, D features, H hidden. Padded to WMMA K=32 tiles.
// ---------------------------------------------------------------------------
constexpr int S  = 4096;
constexpr int M  = 32;
constexpr int D  = 420;
constexpr int H  = 840;
constexpr int Dp = 448;       // 14 * 32
constexpr int Hp = 864;       // 27 * 32
constexpr int KD = Dp / 32;   // 14 k-steps for K=448
constexpr int KH = Hp / 32;   // 27 k-steps for K=864
constexpr int NTB = 7;        // N-tiles per wave block (acc = 56 VGPRs)

typedef __bf16 bf16_t;
typedef __attribute__((ext_vector_type(16))) __bf16 v16bf;
typedef __attribute__((ext_vector_type(8)))  float  v8f;

union FragBits { uint4 u[2]; v16bf v; };
union Pack4    { bf16_t h[4]; uint2 u; };

// Workspace layout (bytes). All segment sizes are 16B multiples.
constexpr size_t oW1   = 0;
constexpr size_t oW2   = oW1  + (size_t)Dp * Dp * 2;
constexpr size_t oW3   = oW2  + (size_t)Dp * Dp * 2;
constexpr size_t oW4   = oW3  + (size_t)Hp * Dp * 2;
constexpr size_t oB1   = oW4  + (size_t)Dp * Hp * 2;
constexpr size_t oB2   = oB1  + (size_t)Dp * 4;
constexpr size_t oB3   = oB2  + (size_t)Dp * 4;
constexpr size_t oB4   = oB3  + (size_t)Hp * 4;
constexpr size_t oPool = oB4  + (size_t)Dp * 4;            // S x Dp bf16
constexpr size_t oY3   = oPool + (size_t)S * Dp * 2;       // S x Hp bf16

// ---------------------------------------------------------------------------
// Fragment loaders for V_WMMA_F32_16X16X32_BF16 (wave32).
// A (16x32 MxK): lane L row m=L&15, half=L>>4; elems j = K[half*16 + j]
//   -> two 16B chunks at k0 and k0+8.
// B (32x16 KxN): lane L col n=L&15, half=L>>4; elems j = K[half*16 + j]
//   -> 32 contiguous bytes of weight row (weights stored [out][in] row-major).
// ---------------------------------------------------------------------------
__device__ __forceinline__ v16bf load_a_frag(const bf16_t* base, int ldk, int kk, int lane) {
  const int m    = lane & 15;
  const int half = lane >> 4;
  const bf16_t* p = base + (size_t)m * ldk + kk * 32 + half * 16;
  FragBits t;
  t.u[0] = *(const uint4*)(p);
  t.u[1] = *(const uint4*)(p + 8);
  return t.v;
}

__device__ __forceinline__ v16bf load_b_frag(const bf16_t* __restrict__ W, int ldk,
                                             int ntile, int kk, int lane) {
  const int n    = ntile * 16 + (lane & 15);
  const int half = lane >> 4;
  const bf16_t* p = W + (size_t)n * ldk + kk * 32 + half * 16;
  FragBits t;
  t.u[0] = *(const uint4*)(p);
  t.u[1] = *(const uint4*)(p + 8);
  return t.v;
}

// ---------------------------------------------------------------------------
// Prep: pad + convert fp32 weights to bf16 [outPad x inPad], zero padding.
// ---------------------------------------------------------------------------
__global__ void k_pad_weight(const float* __restrict__ src, bf16_t* __restrict__ dst,
                             int outReal, int inReal, int inPad, int total) {
  int idx = blockIdx.x * blockDim.x + threadIdx.x;
  if (idx >= total) return;
  int o = idx / inPad;
  int i = idx - o * inPad;
  float v = (o < outReal && i < inReal) ? src[(size_t)o * inReal + i] : 0.0f;
  dst[idx] = (bf16_t)v;
}

__global__ void k_pad_bias(const float* __restrict__ src, float* __restrict__ dst,
                           int real, int pad) {
  int idx = blockIdx.x * blockDim.x + threadIdx.x;
  if (idx < pad) dst[idx] = (idx < real) ? src[idx] : 0.0f;
}

// ---------------------------------------------------------------------------
// Fused layer1 + layer2 + ragged mean-pool. One workgroup per sample:
// 32 rows x 448 cols, 8 waves (one (mtile, 7-ntile-group) job per wave).
// ---------------------------------------------------------------------------
__global__ __launch_bounds__(256) void k_l12_pool(
    const float* __restrict__ X, const int* __restrict__ numInputs,
    const bf16_t* __restrict__ W1, const float* __restrict__ B1,
    const bf16_t* __restrict__ W2, const float* __restrict__ B2,
    bf16_t* __restrict__ pooled) {
  __shared__ bf16_t Xs[M * Dp];    // 28 KB
  __shared__ bf16_t Hs[M * Dp];    // 28 KB
  __shared__ float  Pool[Dp];      // 1.75 KB

  const int s    = blockIdx.x;
  const int tid  = threadIdx.x;
  const int lane = tid & 31;
  const int wave = tid >> 5;
  const int nIn  = numInputs[s];

  // Stage X (fp32 global -> bf16 LDS), vectorized: float4 in, uint2 (4xbf16) out.
  const float* Xrow = X + (size_t)s * M * D;
  constexpr int QROW = D / 4;   // 105 float4 per row
  for (int q = tid; q < M * QROW; q += 256) {
    int r  = q / QROW;
    int cq = q - r * QROW;
    float4 v = ((const float4*)(Xrow + (size_t)r * D))[cq];
    Pack4 p;
    p.h[0] = (bf16_t)v.x; p.h[1] = (bf16_t)v.y;
    p.h[2] = (bf16_t)v.z; p.h[3] = (bf16_t)v.w;
    *(uint2*)&Xs[r * Dp + cq * 4] = p.u;
  }
  for (int idx = tid; idx < M * (Dp - D); idx += 256) {  // zero pad cols
    int r = idx / (Dp - D);
    int c = idx - r * (Dp - D);
    Xs[r * Dp + D + c] = (bf16_t)0.0f;
  }
  for (int idx = tid; idx < Dp; idx += 256) Pool[idx] = 0.0f;
  __syncthreads();

  // Per-wave job: mtile = wave>>2 (0..1), ntile block = (wave&3)*7 (28 ntiles).
  const int mt  = wave >> 2;
  const int nt0 = (wave & 3) * NTB;
  const int cn  = lane & 15;
  const int half = lane >> 4;

  // ---- Layer 1: h1 = relu(X @ W1^T + b1), kept in LDS as bf16 ----
  {
    v8f acc[NTB];
#pragma unroll
    for (int i = 0; i < NTB; ++i) {
      float bias = B1[(nt0 + i) * 16 + cn];
#pragma unroll
      for (int j = 0; j < 8; ++j) acc[i][j] = bias;
    }
#pragma unroll 1
    for (int kk = 0; kk < KD; ++kk) {
      v16bf a = load_a_frag(&Xs[mt * 16 * Dp], Dp, kk, lane);
#pragma unroll
      for (int i = 0; i < NTB; ++i) {
        v16bf b = load_b_frag(W1, Dp, nt0 + i, kk, lane);
        acc[i] = __builtin_amdgcn_wmma_f32_16x16x32_bf16(false, a, false, b,
                                                         (short)0, acc[i], false, false);
      }
    }
#pragma unroll
    for (int i = 0; i < NTB; ++i) {
      const int col = (nt0 + i) * 16 + cn;
#pragma unroll
      for (int v = 0; v < 8; ++v) {
        float r = acc[i][v] > 0.0f ? acc[i][v] : 0.0f;
        Hs[(mt * 16 + v + 8 * half) * Dp + col] = (bf16_t)r;
      }
    }
  }
  __syncthreads();

  // ---- Layer 2 + masked pool: Pool[c] += sum_{m < nIn} relu(h1@W2^T + b2) ----
  {
    v8f acc[NTB];
#pragma unroll
    for (int i = 0; i < NTB; ++i) {
      float bias = B2[(nt0 + i) * 16 + cn];
#pragma unroll
      for (int j = 0; j < 8; ++j) acc[i][j] = bias;
    }
#pragma unroll 1
    for (int kk = 0; kk < KD; ++kk) {
      v16bf a = load_a_frag(&Hs[mt * 16 * Dp], Dp, kk, lane);
#pragma unroll
      for (int i = 0; i < NTB; ++i) {
        v16bf b = load_b_frag(W2, Dp, nt0 + i, kk, lane);
        acc[i] = __builtin_amdgcn_wmma_f32_16x16x32_bf16(false, a, false, b,
                                                         (short)0, acc[i], false, false);
      }
    }
#pragma unroll
    for (int i = 0; i < NTB; ++i) {
      const int col = (nt0 + i) * 16 + cn;
      float partial = 0.0f;
#pragma unroll
      for (int v = 0; v < 8; ++v) {
        float r = acc[i][v] > 0.0f ? acc[i][v] : 0.0f;
        int mrow = mt * 16 + v + 8 * half;   // input-slot index within sample
        if (mrow < nIn) partial += r;
      }
      atomicAdd(&Pool[col], partial);        // ds_add_f32
    }
  }
  __syncthreads();

  const float inv = 1.0f / (float)nIn;
  for (int idx = tid; idx < Dp; idx += 256)
    pooled[(size_t)s * Dp + idx] = (bf16_t)(Pool[idx] * inv);
}

// ---------------------------------------------------------------------------
// Layer 3: y3 = relu(pooled @ W3^T + b3) -> bf16 [S x Hp].
// 16-row tiles, 8 waves, 7-ntile blocks (54 = 7*7 + 5; wave-uniform remainder).
// ---------------------------------------------------------------------------
__global__ __launch_bounds__(256) void k_l3(
    const bf16_t* __restrict__ pooled, const bf16_t* __restrict__ W3,
    const float* __restrict__ B3, bf16_t* __restrict__ y3) {
  __shared__ bf16_t As[16 * Dp];   // 14 KB
  const int rb   = blockIdx.x * 16;
  const int tid  = threadIdx.x;
  const int lane = tid & 31;
  const int wave = tid >> 5;
  const int cn   = lane & 15;
  const int half = lane >> 4;

  const uint4* src = (const uint4*)(pooled + (size_t)rb * Dp);
  uint4* dst = (uint4*)As;
  for (int idx = tid; idx < 16 * Dp / 8; idx += 256) dst[idx] = src[idx];
  __syncthreads();

  const int nt0 = wave * NTB;                 // 8 groups of 7 -> covers 54 (+2 pad)
  const int cnt = min(NTB, Hp / 16 - nt0);    // wave-uniform: 7,...,7,5

  v8f acc[NTB];
#pragma unroll
  for (int i = 0; i < NTB; ++i) {
    if (i < cnt) {
      float bias = B3[(nt0 + i) * 16 + cn];
#pragma unroll
      for (int j = 0; j < 8; ++j) acc[i][j] = bias;
    }
  }
#pragma unroll 1
  for (int kk = 0; kk < KD; ++kk) {
    v16bf a = load_a_frag(As, Dp, kk, lane);
#pragma unroll
    for (int i = 0; i < NTB; ++i) {
      if (i < cnt) {
        v16bf b = load_b_frag(W3, Dp, nt0 + i, kk, lane);
        acc[i] = __builtin_amdgcn_wmma_f32_16x16x32_bf16(false, a, false, b,
                                                         (short)0, acc[i], false, false);
      }
    }
  }
#pragma unroll
  for (int i = 0; i < NTB; ++i) {
    if (i < cnt) {
      const int col = (nt0 + i) * 16 + cn;
#pragma unroll
      for (int v = 0; v < 8; ++v) {
        float r = acc[i][v] > 0.0f ? acc[i][v] : 0.0f;
        y3[(size_t)(rb + v + 8 * half) * Hp + col] = (bf16_t)r;
      }
    }
  }
}

// ---------------------------------------------------------------------------
// Layer 4: out = relu(y3 @ W4^T + b4) -> f32 [S x D] (pad cols dropped).
// 16-row tiles, 4 waves x 7-ntile blocks (28 ntiles exactly).
// ---------------------------------------------------------------------------
__global__ __launch_bounds__(128) void k_l4(
    const bf16_t* __restrict__ y3, const bf16_t* __restrict__ W4,
    const float* __restrict__ B4, float* __restrict__ out) {
  __shared__ bf16_t As[16 * Hp];   // 27 KB
  const int rb   = blockIdx.x * 16;
  const int tid  = threadIdx.x;
  const int lane = tid & 31;
  const int wave = tid >> 5;
  const int cn   = lane & 15;
  const int half = lane >> 4;

  const uint4* src = (const uint4*)(y3 + (size_t)rb * Hp);
  uint4* dst = (uint4*)As;
  for (int idx = tid; idx < 16 * Hp / 8; idx += 128) dst[idx] = src[idx];
  __syncthreads();

  const int nt0 = wave * NTB;   // 4 waves x 7 = 28 ntiles

  v8f acc[NTB];
#pragma unroll
  for (int i = 0; i < NTB; ++i) {
    float bias = B4[(nt0 + i) * 16 + cn];
#pragma unroll
    for (int j = 0; j < 8; ++j) acc[i][j] = bias;
  }
#pragma unroll 1
  for (int kk = 0; kk < KH; ++kk) {
    v16bf a = load_a_frag(As, Hp, kk, lane);
#pragma unroll
    for (int i = 0; i < NTB; ++i) {
      v16bf b = load_b_frag(W4, Hp, nt0 + i, kk, lane);
      acc[i] = __builtin_amdgcn_wmma_f32_16x16x32_bf16(false, a, false, b,
                                                       (short)0, acc[i], false, false);
    }
  }
#pragma unroll
  for (int i = 0; i < NTB; ++i) {
    const int col = (nt0 + i) * 16 + cn;
    if (col < D) {
#pragma unroll
      for (int v = 0; v < 8; ++v) {
        float r = acc[i][v] > 0.0f ? acc[i][v] : 0.0f;
        out[(size_t)(rb + v + 8 * half) * D + col] = r;
      }
    }
  }
}

// ---------------------------------------------------------------------------
extern "C" void kernel_launch(void* const* d_in, const int* in_sizes, int n_in,
                              void* d_out, int out_size, void* d_ws, size_t ws_size,
                              hipStream_t stream) {
  const float* all_x = (const float*)d_in[0];
  const int*   nIn   = (const int*)d_in[1];
  const float* W1 = (const float*)d_in[2];
  const float* b1 = (const float*)d_in[3];
  const float* W2 = (const float*)d_in[4];
  const float* b2 = (const float*)d_in[5];
  const float* W3 = (const float*)d_in[6];
  const float* b3 = (const float*)d_in[7];
  const float* W4 = (const float*)d_in[8];
  const float* b4 = (const float*)d_in[9];
  float* out = (float*)d_out;

  char* ws = (char*)d_ws;
  bf16_t* W1b = (bf16_t*)(ws + oW1);
  bf16_t* W2b = (bf16_t*)(ws + oW2);
  bf16_t* W3b = (bf16_t*)(ws + oW3);
  bf16_t* W4b = (bf16_t*)(ws + oW4);
  float*  B1f = (float*)(ws + oB1);
  float*  B2f = (float*)(ws + oB2);
  float*  B3f = (float*)(ws + oB3);
  float*  B4f = (float*)(ws + oB4);
  bf16_t* pooledB = (bf16_t*)(ws + oPool);
  bf16_t* y3B     = (bf16_t*)(ws + oY3);

  // Weight/bias prep (tiny; ~2.3 MB written once per launch).
  k_pad_weight<<<(Dp * Dp + 255) / 256, 256, 0, stream>>>(W1, W1b, D, D, Dp, Dp * Dp);
  k_pad_weight<<<(Dp * Dp + 255) / 256, 256, 0, stream>>>(W2, W2b, D, D, Dp, Dp * Dp);
  k_pad_weight<<<(Hp * Dp + 255) / 256, 256, 0, stream>>>(W3, W3b, H, D, Dp, Hp * Dp);
  k_pad_weight<<<(Dp * Hp + 255) / 256, 256, 0, stream>>>(W4, W4b, D, H, Hp, Dp * Hp);
  k_pad_bias<<<(Dp + 255) / 256, 256, 0, stream>>>(b1, B1f, D, Dp);
  k_pad_bias<<<(Dp + 255) / 256, 256, 0, stream>>>(b2, B2f, D, Dp);
  k_pad_bias<<<(Hp + 255) / 256, 256, 0, stream>>>(b3, B3f, H, Hp);
  k_pad_bias<<<(Dp + 255) / 256, 256, 0, stream>>>(b4, B4f, D, Dp);

  // Fused L1+L2+pool: one workgroup per sample (4096 WGs x 8 waves).
  k_l12_pool<<<S, 256, 0, stream>>>(all_x, nIn, W1b, B1f, W2b, B2f, pooledB);
  // Post-MLP.
  k_l3<<<S / 16, 256, 0, stream>>>(pooledB, W3b, B3f, y3B);
  k_l4<<<S / 16, 128, 0, stream>>>(y3B, W4b, B4f, out);
}
